// GCN2_12893491822964
// MI455X (gfx1250) — compile-verified, matrix operand order checked by reading, not run on hardware
//
#include <hip/hip_runtime.h>
#include <cmath>

#define IN_CH 128
#define HID   64
#define OUT_CH 40
#define NUM_LAYERS 8
#define ALPHA 0.1f

typedef __attribute__((ext_vector_type(16))) __bf16        v16bf;
typedef __attribute__((ext_vector_type(8)))  float         v8f;
typedef __attribute__((ext_vector_type(4)))  unsigned int  v4u;

union BF16Frag {
    v16bf          v;
    v4u            q[2];
    unsigned short u[16];
};

// fp32 -> bf16 (round-to-nearest-even)
__device__ __forceinline__ unsigned short f2bf(float f) {
    unsigned int u = __float_as_uint(f);
    u += 0x7FFFu + ((u >> 16) & 1u);
    return (unsigned short)(u >> 16);
}

// Fragment-packed layout [t][s][lane][e] -> (k, col) of the weight element.
__device__ __forceinline__ void frag_coords(int L, int e, int s, int t,
                                            int* k, int* col) {
    int khalf = (L >> 4) * 8;                                // 0 or 8
    int kin   = (e < 8) ? (khalf + e) : (khalf + 16 + (e - 8));
    *k   = s * 32 + kin;
    *col = t * 16 + (L & 15);
}

// ---------------------------------------------------------------------------
// Kernel 1: x0 = relu(x @ W_in + b_in); also h = x0.
// Block = 16 rows x 64 cols, 4 waves, each wave owns one 16x16 N-tile.
// ---------------------------------------------------------------------------
__global__ __launch_bounds__(128) void entry_kernel(
    const float* __restrict__ x, const float* __restrict__ W_in,
    const float* __restrict__ b_in, float* __restrict__ x0,
    float* __restrict__ h, int nrows)
{
    __shared__ __align__(16) unsigned short As[16][IN_CH + 8];
    __shared__ __align__(16) unsigned short Bp[4][4][32][16]; // [t][s][lane][e]

    const int tid = threadIdx.x;
    const int row_base = blockIdx.x * 16;
    const bool full = (row_base + 16 <= nrows);

    // W_in (128x64) -> LDS, pre-packed in B-fragment order.
    unsigned short* bp = &Bp[0][0][0][0];
    for (int i = tid; i < 4 * 4 * 32 * 16; i += 128) {
        int e = i & 15, L = (i >> 4) & 31, s = (i >> 9) & 3, t = i >> 11;
        int k, col;
        frag_coords(L, e, s, t, &k, &col);
        bp[i] = f2bf(W_in[k * HID + col]);
    }
    // A tile (16x128) -> LDS (row-major; lane reads are K-contiguous).
    if (full) {
        for (int i = tid; i < 16 * IN_CH; i += 128) {
            int r = i / IN_CH, k = i % IN_CH;
            As[r][k] = f2bf(x[(size_t)(row_base + r) * IN_CH + k]);
        }
    } else {
        for (int i = tid; i < 16 * IN_CH; i += 128) {
            int r = i / IN_CH, k = i % IN_CH;
            int rg = row_base + r;
            As[r][k] = (rg < nrows) ? f2bf(x[(size_t)rg * IN_CH + k])
                                    : (unsigned short)0;
        }
    }
    __syncthreads();

    const int lane  = tid & 31;
    const int wave  = tid >> 5;
    const int col   = wave * 16 + (lane & 15);
    const int khalf = (lane >> 4) * 8;
    const int arow  = lane & 15;

    v8f acc = {};
#pragma unroll
    for (int ks = 0; ks < 4; ++ks) {
        const int kk = ks * 32;
        BF16Frag a, b;
        a.q[0] = *reinterpret_cast<const v4u*>(&As[arow][kk + khalf]);
        a.q[1] = *reinterpret_cast<const v4u*>(&As[arow][kk + khalf + 16]);
        const v4u* bq = reinterpret_cast<const v4u*>(Bp[wave][ks][lane]);
        b.q[0] = bq[0];
        b.q[1] = bq[1];
        acc = __builtin_amdgcn_wmma_f32_16x16x32_bf16(
            false, a.v, false, b.v, (short)0, acc, false, false);
    }

    const float bias = b_in[col];
    const int rtop = (lane >> 4) * 8;
    if (full) {
#pragma unroll
        for (int v = 0; v < 8; ++v) {
            int r = row_base + rtop + v;
            float val = acc[v] + bias;
            val = val > 0.f ? val : 0.f;
            x0[r * HID + col] = val;
            h[r * HID + col]  = val;
        }
    } else {
#pragma unroll
        for (int v = 0; v < 8; ++v) {
            int r = row_base + rtop + v;
            if (r < nrows) {
                float val = acc[v] + bias;
                val = val > 0.f ? val : 0.f;
                x0[r * HID + col] = val;
                h[r * HID + col]  = val;
            }
        }
    }
}

// ---------------------------------------------------------------------------
// Kernel 2: zero agg
// ---------------------------------------------------------------------------
__global__ __launch_bounds__(256) void zero_kernel(float4* __restrict__ p, int n4)
{
    int i = blockIdx.x * 256 + threadIdx.x;
    if (i < n4) p[i] = make_float4(0.f, 0.f, 0.f, 0.f);
}

// ---------------------------------------------------------------------------
// Kernel 3: SpMM scatter: agg[dst] += w * h[src]
// One thread per (edge, 4-channel group): float4 gather + 4 f32 atomics (L2).
// ---------------------------------------------------------------------------
__global__ __launch_bounds__(256) void spmm_kernel(
    const int* __restrict__ esrc, const int* __restrict__ edst,
    const float* __restrict__ ew, const float* __restrict__ h,
    float* __restrict__ agg, int nedges)
{
    int idx = blockIdx.x * 256 + threadIdx.x;
    if (idx >= nedges * (HID / 4)) return;
    int e = idx >> 4;
    int c = (idx & 15) << 2;

    // speculative prefetch ~4K edges ahead (one lane per edge does it)
    int epf = e + 4096;
    if ((idx & 15) == 0 && epf < nedges) {
        __builtin_prefetch(h + (size_t)esrc[epf] * HID, 0, 0);
    }

    int s = esrc[e], d = edst[e];
    float wt = ew[e];
    const float4 v = *reinterpret_cast<const float4*>(h + s * HID + c);
    float* o = agg + d * HID + c;
    atomicAdd(o + 0, v.x * wt);
    atomicAdd(o + 1, v.y * wt);
    atomicAdd(o + 2, v.z * wt);
    atomicAdd(o + 3, v.w * wt);
}

// ---------------------------------------------------------------------------
// Kernel 4: fused layer: u = (1-a)*agg + a*x0 ; h = relu((1-b)*u + b*(u@Wl))
// ---------------------------------------------------------------------------
__global__ __launch_bounds__(128) void layer_kernel(
    const float* __restrict__ agg, const float* __restrict__ x0,
    const float* __restrict__ Wl, float* __restrict__ h,
    float beta, int nrows)
{
    __shared__ __align__(16) float          Uf[16][HID];
    __shared__ __align__(16) unsigned short Ub[16][HID + 8];
    __shared__ __align__(16) unsigned short Bp[4][2][32][16]; // [t][s][lane][e]

    const int tid = threadIdx.x;
    const int row_base = blockIdx.x * 16;
    const bool full = (row_base + 16 <= nrows);

    unsigned short* bp = &Bp[0][0][0][0];
    for (int i = tid; i < 4 * 2 * 32 * 16; i += 128) {
        int e = i & 15, L = (i >> 4) & 31, s = (i >> 9) & 1, t = i >> 10;
        int k, col;
        frag_coords(L, e, s, t, &k, &col);
        bp[i] = f2bf(Wl[k * HID + col]);
    }
    if (full) {
        for (int i = tid; i < 16 * HID; i += 128) {
            int r = i / HID, c = i % HID;
            int rg = row_base + r;
            float u = (1.0f - ALPHA) * agg[rg * HID + c] + ALPHA * x0[rg * HID + c];
            Uf[r][c] = u;
            Ub[r][c] = f2bf(u);
        }
    } else {
        for (int i = tid; i < 16 * HID; i += 128) {
            int r = i / HID, c = i % HID;
            int rg = row_base + r;
            float u = 0.f;
            if (rg < nrows)
                u = (1.0f - ALPHA) * agg[rg * HID + c] + ALPHA * x0[rg * HID + c];
            Uf[r][c] = u;
            Ub[r][c] = f2bf(u);
        }
    }
    __syncthreads();

    const int lane  = tid & 31;
    const int wave  = tid >> 5;
    const int col   = wave * 16 + (lane & 15);
    const int khalf = (lane >> 4) * 8;
    const int arow  = lane & 15;

    v8f acc = {};
#pragma unroll
    for (int ks = 0; ks < 2; ++ks) {
        const int kk = ks * 32;
        BF16Frag a, b;
        a.q[0] = *reinterpret_cast<const v4u*>(&Ub[arow][kk + khalf]);
        a.q[1] = *reinterpret_cast<const v4u*>(&Ub[arow][kk + khalf + 16]);
        const v4u* bq = reinterpret_cast<const v4u*>(Bp[wave][ks][lane]);
        b.q[0] = bq[0];
        b.q[1] = bq[1];
        acc = __builtin_amdgcn_wmma_f32_16x16x32_bf16(
            false, a.v, false, b.v, (short)0, acc, false, false);
    }

    const int rtop = (lane >> 4) * 8;
    if (full) {
#pragma unroll
        for (int v = 0; v < 8; ++v) {
            int r = rtop + v;
            float u = Uf[r][col];
            float val = (1.0f - beta) * u + beta * acc[v];
            val = val > 0.f ? val : 0.f;
            h[(row_base + r) * HID + col] = val;
        }
    } else {
#pragma unroll
        for (int v = 0; v < 8; ++v) {
            int r = rtop + v;
            int rg = row_base + r;
            if (rg < nrows) {
                float u = Uf[r][col];
                float val = (1.0f - beta) * u + beta * acc[v];
                val = val > 0.f ? val : 0.f;
                h[rg * HID + col] = val;
            }
        }
    }
}

// ---------------------------------------------------------------------------
// Kernel 5: out = h @ W_out + b_out  (W_out zero-padded to 64 cols in LDS so
// all 4 waves run WMMA with full EXEC; masked store of the 40 real columns)
// ---------------------------------------------------------------------------
__global__ __launch_bounds__(128) void out_kernel(
    const float* __restrict__ h, const float* __restrict__ W_out,
    const float* __restrict__ b_out, float* __restrict__ out, int nrows)
{
    __shared__ __align__(16) unsigned short Hs[16][HID + 8];
    __shared__ __align__(16) unsigned short Bp[4][2][32][16]; // [t][s][lane][e]

    const int tid = threadIdx.x;
    const int row_base = blockIdx.x * 16;
    const bool full = (row_base + 16 <= nrows);

    unsigned short* bp = &Bp[0][0][0][0];
    for (int i = tid; i < 4 * 2 * 32 * 16; i += 128) {
        int e = i & 15, L = (i >> 4) & 31, s = (i >> 9) & 1, t = i >> 10;
        int k, col;
        frag_coords(L, e, s, t, &k, &col);
        bp[i] = (col < OUT_CH) ? f2bf(W_out[k * OUT_CH + col]) : (unsigned short)0;
    }
    if (full) {
        for (int i = tid; i < 16 * HID; i += 128) {
            int r = i / HID, k = i % HID;
            Hs[r][k] = f2bf(h[(row_base + r) * HID + k]);
        }
    } else {
        for (int i = tid; i < 16 * HID; i += 128) {
            int r = i / HID, k = i % HID;
            int rg = row_base + r;
            Hs[r][k] = (rg < nrows) ? f2bf(h[rg * HID + k]) : (unsigned short)0;
        }
    }
    __syncthreads();

    const int lane  = tid & 31;
    const int wave  = tid >> 5;
    const int col   = wave * 16 + (lane & 15);
    const int khalf = (lane >> 4) * 8;
    const int arow  = lane & 15;

    v8f acc = {};
#pragma unroll
    for (int ks = 0; ks < 2; ++ks) {
        const int kk = ks * 32;
        BF16Frag a, b;
        a.q[0] = *reinterpret_cast<const v4u*>(&Hs[arow][kk + khalf]);
        a.q[1] = *reinterpret_cast<const v4u*>(&Hs[arow][kk + khalf + 16]);
        const v4u* bq = reinterpret_cast<const v4u*>(Bp[wave][ks][lane]);
        b.q[0] = bq[0];
        b.q[1] = bq[1];
        acc = __builtin_amdgcn_wmma_f32_16x16x32_bf16(
            false, a.v, false, b.v, (short)0, acc, false, false);
    }

    if (col < OUT_CH) {
        const float bias = b_out[col];
        const int rtop = (lane >> 4) * 8;
        if (full) {
#pragma unroll
            for (int v = 0; v < 8; ++v) {
                out[(row_base + rtop + v) * OUT_CH + col] = acc[v] + bias;
            }
        } else {
#pragma unroll
            for (int v = 0; v < 8; ++v) {
                int rg = row_base + rtop + v;
                if (rg < nrows) out[rg * OUT_CH + col] = acc[v] + bias;
            }
        }
    }
}

// ---------------------------------------------------------------------------
extern "C" void kernel_launch(void* const* d_in, const int* in_sizes, int n_in,
                              void* d_out, int out_size, void* d_ws, size_t ws_size,
                              hipStream_t stream)
{
    const float* x       = (const float*)d_in[0];
    const int*   esrc    = (const int*)  d_in[1];
    const int*   edst    = (const int*)  d_in[2];
    const float* ew      = (const float*)d_in[3];
    const float* W_in    = (const float*)d_in[4];
    const float* b_in    = (const float*)d_in[5];
    const float* W_convs = (const float*)d_in[6];
    const float* W_out   = (const float*)d_in[7];
    const float* b_out   = (const float*)d_in[8];

    const int Nn = in_sizes[0] / IN_CH;   // 50000
    const int Ee = in_sizes[1];           // 800000

    float* x0  = (float*)d_ws;
    float* h   = x0 + (size_t)Nn * HID;
    float* agg = h  + (size_t)Nn * HID;

    const int mblocks = (Nn + 15) / 16;

    entry_kernel<<<mblocks, 128, 0, stream>>>(x, W_in, b_in, x0, h, Nn);

    const int n4 = Nn * HID / 4;
    const int zb = (n4 + 255) / 256;
    const int sthreads = Ee * (HID / 4);
    const int sb = (sthreads + 255) / 256;

    for (int l = 0; l < NUM_LAYERS; ++l) {
        zero_kernel<<<zb, 256, 0, stream>>>((float4*)agg, n4);
        spmm_kernel<<<sb, 256, 0, stream>>>(esrc, edst, ew, h, agg, Ee);
        float beta = logf(0.5f / (float)(l + 1) + 1.0f);
        layer_kernel<<<mblocks, 128, 0, stream>>>(
            agg, x0, W_convs + (size_t)l * HID * HID, h, beta, Nn);
    }

    out_kernel<<<mblocks, 128, 0, stream>>>(h, W_out, b_out, (float*)d_out, Nn);
}
